// GINOTGeometryEncoder_3032246911133
// MI455X (gfx1250) — matrix operand order repaired and checked by reading, not compiled, not run on hardware
//
#include <hip/hip_runtime.h>
#include <math.h>

// ---------------- constants ----------------
#define BB 8
#define NN 4096
#define NS 1024
#define NP 32
#define DM 128
#define NH 4
#define DH 32
#define FF 512
#define MLPH 256
#define LN_EPS 1e-5f

typedef _Float16 v16h __attribute__((ext_vector_type(16)));
typedef _Float16 v8h  __attribute__((ext_vector_type(8)));
typedef float    v8f  __attribute__((ext_vector_type(8)));

#define DEV static __device__ __forceinline__

// ---- WMMA fragment helpers (layouts per CDNA5 ISA 7.12.2, wave32) ----
// A (16x32 f16): lane<16 holds row M=lane, K = {0..7 in v0..3, 16..23 in v4..7};
// lane>=16 same row, K = {8..15, 24..31}.  => K(e) = half*8 + (e&7) + (e&8)*2
DEV v16h load_a_frag(const _Float16* base, int lda, int lane) {
  const _Float16* p = base + (size_t)(lane & 15) * lda + ((lane >> 4) << 3);
  v8h lo = *(const v8h*)p;
  v8h hi = *(const v8h*)(p + 16);
  v16h a;
#pragma unroll
  for (int i = 0; i < 8; ++i) { a[i] = lo[i]; a[i + 8] = hi[i]; }
  return a;
}
// B (32x16 f16): lane<16 holds col N=lane, K = {0..7, 16..23}; lane>=16 K = {8..15, 24..31}.
// B[k][n] = W[nBase+n][kBase+k] for C = A @ W^T with W row-major (N x K).
DEV v16h load_b_frag(const _Float16* wrow0, int ldw, int lane) {
  const _Float16* p = wrow0 + (size_t)(lane & 15) * ldw + ((lane >> 4) << 3);
  v8h lo = *(const v8h*)p;
  v8h hi = *(const v8h*)(p + 16);
  v16h b;
#pragma unroll
  for (int i = 0; i < 8; ++i) { b[i] = lo[i]; b[i + 8] = hi[i]; }
  return b;
}
DEV v8f wmma16(v16h a, v16h b, v8f c) {
  return __builtin_amdgcn_wmma_f32_16x16x32_f16(false, a, false, b, (short)0, c,
                                                false, false);
}
DEV float gelu_f(float x) { return 0.5f * x * (1.0f + erff(x * 0.70710678118654752f)); }

// ---------------- weight conversion ----------------
__global__ void k_cvt(const float* __restrict__ s, _Float16* __restrict__ d, int n) {
  for (int i = blockIdx.x * blockDim.x + threadIdx.x; i < n; i += gridDim.x * blockDim.x)
    d[i] = (_Float16)s[i];
}
__global__ void k_cvt_pad(const float* __restrict__ s, _Float16* __restrict__ d,
                          int rows, int cols, int ld) {
  int n = rows * ld;
  for (int i = blockIdx.x * blockDim.x + threadIdx.x; i < n; i += gridDim.x * blockDim.x) {
    int r = i / ld, c = i - r * ld;
    d[i] = (c < cols) ? (_Float16)s[r * cols + c] : (_Float16)0.0f;
  }
}

// ---------------- fourier features + gfeat ----------------
__global__ void k_gfeat(const float* __restrict__ pts, const float* __restrict__ fw,
                        const float* __restrict__ fb, _Float16* __restrict__ g16) {
  int token = blockIdx.x;           // B*N tokens
  int t = threadIdx.x;              // 128 channels
  __shared__ float f[26];
  float x0 = pts[token * 2 + 0], x1 = pts[token * 2 + 1];
  if (t < 26) {
    float v;
    if (t < 2) v = (t == 0) ? x0 : x1;
    else {
      int j = t - 2, k = j >> 2, r = j & 3;
      float w = (float)(1 << k) * 3.14159265358979323846f;
      float a = w * ((r & 1) ? x1 : x0);
      v = (r < 2) ? sinf(a) : cosf(a);
    }
    f[t] = v;
  }
  __syncthreads();
  float acc = fb[t];
#pragma unroll
  for (int k = 0; k < 26; ++k) acc += fw[t * 26 + k] * f[k];
  g16[(size_t)token * DM + t] = (_Float16)acc;
}

// ---------------- farthest point sampling (1 block / batch) ----------------
__global__ __launch_bounds__(256) void k_fps(const float* __restrict__ pts,
                                             int* __restrict__ fidx,
                                             float* __restrict__ cent) {
  int b = blockIdx.x, tid = threadIdx.x;
  __shared__ float sx[NN], sy[NN], sd[NN];
  __shared__ float rv[256];
  __shared__ int   ri[256];
  __shared__ int   s_last;
  for (int i = tid; i < NN; i += 256) {
    sx[i] = pts[((size_t)b * NN + i) * 2 + 0];
    sy[i] = pts[((size_t)b * NN + i) * 2 + 1];
    sd[i] = __builtin_inff();
  }
  __syncthreads();
  if (tid == 0) {
    fidx[b * NS] = 0;
    cent[(size_t)(b * NS) * 2 + 0] = sx[0];
    cent[(size_t)(b * NS) * 2 + 1] = sy[0];
    s_last = 0;
  }
  __syncthreads();
  for (int i = 1; i < NS; ++i) {
    int last = s_last;
    float lx = sx[last], ly = sy[last];
    float best = -1.0f; int bi = 0;
    for (int j = tid; j < NN; j += 256) {
      float dx = sx[j] - lx, dy = sy[j] - ly;
      float d = fminf(sd[j], dx * dx + dy * dy);
      sd[j] = d;
      if (d > best) { best = d; bi = j; }
    }
    rv[tid] = best; ri[tid] = bi;
    __syncthreads();
    for (int s = 128; s > 0; s >>= 1) {
      if (tid < s) {
        float ov = rv[tid + s]; int oi = ri[tid + s];
        if (ov > rv[tid] || (ov == rv[tid] && oi < ri[tid])) { rv[tid] = ov; ri[tid] = oi; }
      }
      __syncthreads();
    }
    if (tid == 0) {
      int nx = ri[0];
      s_last = nx;
      fidx[b * NS + i] = nx;
      cent[(size_t)(b * NS + i) * 2 + 0] = sx[nx];
      cent[(size_t)(b * NS + i) * 2 + 1] = sy[nx];
    }
    __syncthreads();
  }
}

// ---------------- ball query: first 32 indices (ascending) within radius ----------------
__global__ __launch_bounds__(256) void k_ballq(const float* __restrict__ pts,
                                               const float* __restrict__ cent,
                                               int* __restrict__ knn,
                                               int* __restrict__ cnts) {
  int b = blockIdx.y, tid = threadIdx.x;
  int g = b * NS + blockIdx.x * 256 + tid;
  float cx = cent[(size_t)g * 2 + 0], cy = cent[(size_t)g * 2 + 1];
  __shared__ float px[1024], py[1024];
  int cnt = 0, base = g * NP;
  for (int chunk = 0; chunk < NN / 1024; ++chunk) {
    int off = chunk * 1024;
    for (int j = tid; j < 1024; j += 256) {
      px[j] = pts[((size_t)b * NN + off + j) * 2 + 0];
      py[j] = pts[((size_t)b * NN + off + j) * 2 + 1];
    }
    __syncthreads();
    for (int j = 0; j < 1024; ++j) {
      float dx = px[j] - cx, dy = py[j] - cy;
      if (dx * dx + dy * dy < 0.01f && cnt < NP) { knn[base + cnt] = off + j; ++cnt; }
    }
    __syncthreads();
  }
  int first = (cnt > 0) ? knn[base] : (NN - 1);
  for (int k = cnt; k < NP; ++k) knn[base + k] = first;
  cnts[g] = cnt;
}

// ---------------- grouped MLP (1 block / centroid; 32 rows) ----------------
DEV void block_gemm(const _Float16* aBuf, const _Float16* W, const float* bias,
                    float* cBuf, int K, int N, int ldw, int tid) {
  int lane = tid & 31, wave = tid >> 5;
  int tilesTotal = 2 * (N >> 4);
  int perWave = tilesTotal >> 3;
  for (int i = 0; i < perWave; ++i) {
    int t = wave * perWave + i;
    int mT = t & 1, nT = t >> 1;
    v8f acc = {};
    for (int kc = 0; kc < K; kc += 32) {
      v16h a = load_a_frag(aBuf + mT * 16 * 256 + kc, 256, lane);
      v16h b = load_b_frag(W + (size_t)(nT * 16) * ldw + kc, ldw, lane);
      acc = wmma16(a, b, acc);
    }
    int n = lane & 15, half = lane >> 4;
    int col = nT * 16 + n;
    float bv = bias[col];
#pragma unroll
    for (int e = 0; e < 8; ++e) {
      int row = mT * 16 + e + (half << 3);
      cBuf[row * 256 + col] = acc[e] + bv;
    }
  }
}
DEV void ln_gelu_block(const float* cBuf, _Float16* aBuf, const float* g,
                       const float* be, float* rM, float* rR, int tid) {
  if (tid < 32) {
    float s1 = 0.f, s2 = 0.f;
    for (int c = 0; c < 256; ++c) { float v = cBuf[tid * 256 + c]; s1 += v; s2 += v * v; }
    float m = s1 * (1.0f / 256.0f);
    rM[tid] = m;
    rR[tid] = rsqrtf(s2 * (1.0f / 256.0f) - m * m + LN_EPS);
  }
  __syncthreads();
  for (int e = tid; e < 32 * 256; e += 256) {
    int r = e >> 8, c = e & 255;
    float v = (cBuf[e] - rM[r]) * rR[r] * g[c] + be[c];
    aBuf[e] = (_Float16)gelu_f(v);
  }
  __syncthreads();
}

__global__ __launch_bounds__(256) void k_groupmlp(
    const float* __restrict__ pts, const _Float16* __restrict__ g16,
    const int* __restrict__ knn, const int* __restrict__ cnts,
    const float* __restrict__ cent,
    const _Float16* __restrict__ w1p, const float* __restrict__ b1,
    const float* __restrict__ g1, const float* __restrict__ be1,
    const _Float16* __restrict__ w2h, const float* __restrict__ b2,
    const float* __restrict__ g2, const float* __restrict__ be2,
    const _Float16* __restrict__ w3h, const float* __restrict__ b3,
    float* __restrict__ x32, _Float16* __restrict__ x16) {
  __shared__ __align__(32) _Float16 aBuf[32 * 256];
  __shared__ __align__(32) float    cBuf[32 * 256];
  __shared__ float rM[32], rR[32], sRx[32], sRy[32];
  __shared__ int   sIdx[32];
  __shared__ int   sCnt;
  int tid = threadIdx.x;
  int b = blockIdx.x >> 10, s = blockIdx.x & (NS - 1);
  if (tid < 32) {
    int id = knn[(size_t)(b * NS + s) * NP + tid];
    sIdx[tid] = id;
    float cx = cent[(size_t)(b * NS + s) * 2 + 0];
    float cy = cent[(size_t)(b * NS + s) * 2 + 1];
    sRx[tid] = pts[((size_t)b * NN + id) * 2 + 0] - cx;
    sRy[tid] = pts[((size_t)b * NN + id) * 2 + 1] - cy;
  }
  if (tid == 0) sCnt = cnts[b * NS + s];
  __syncthreads();
  // gin: [feat(128) | rel(2) | pad -> 160], stored with row stride 256
  for (int e = tid; e < 32 * 256; e += 256) {
    int r = e >> 8, c = e & 255;
    _Float16 v;
    if (c < 128)       v = g16[((size_t)b * NN + sIdx[r]) * DM + c];
    else if (c == 128) v = (_Float16)sRx[r];
    else if (c == 129) v = (_Float16)sRy[r];
    else               v = (_Float16)0.0f;
    aBuf[e] = v;
  }
  __syncthreads();
  block_gemm(aBuf, w1p, b1, cBuf, 160, 256, 160, tid);
  __syncthreads();
  ln_gelu_block(cBuf, aBuf, g1, be1, rM, rR, tid);
  block_gemm(aBuf, w2h, b2, cBuf, 256, 256, 256, tid);
  __syncthreads();
  ln_gelu_block(cBuf, aBuf, g2, be2, rM, rR, tid);
  block_gemm(aBuf, w3h, b3, cBuf, 256, 128, 256, tid);
  __syncthreads();
  if (tid < 128) {
    int cn = sCnt;
    float m = -__builtin_inff();
    for (int r = 0; r < cn; ++r) m = fmaxf(m, cBuf[r * 256 + tid]);
    if (cn <= 0) m = 0.0f;
    x32[(size_t)(b * NS + s) * DM + tid] = m;
    x16[(size_t)(b * NS + s) * DM + tid] = (_Float16)m;
  }
}

// ---------------- head projection (Q/K normal, V transposed) ----------------
__global__ __launch_bounds__(256) void k_proj(const _Float16* __restrict__ src,
                                              const _Float16* __restrict__ W,
                                              const float* __restrict__ bias,
                                              _Float16* __restrict__ dst,
                                              int S, int vtrans) {
  int tid = threadIdx.x, lane = tid & 31, wave = tid >> 5;
  int rowBase = blockIdx.x * 32;
  for (int i = 0; i < 2; ++i) {
    int t = wave * 2 + i, mT = t & 1, nT = t >> 1;
    v8f acc = {};
    for (int kc = 0; kc < DM; kc += 32) {
      v16h a = load_a_frag(src + (size_t)(rowBase + mT * 16) * DM + kc, DM, lane);
      v16h b = load_b_frag(W + (size_t)(nT * 16) * DM + kc, DM, lane);
      acc = wmma16(a, b, acc);
    }
    int n = lane & 15, half = lane >> 4;
    int col = nT * 16 + n;
    float bv = bias[col];
    int h = col >> 5, dh = col & 31;
#pragma unroll
    for (int e = 0; e < 8; ++e) {
      int m = rowBase + mT * 16 + e + (half << 3);
      int bb = m / S, tt = m - bb * S;
      float v = acc[e] + bv;
      size_t idx = vtrans ? ((size_t)((bb * NH + h) * DH + dh) * S + tt)
                          : (((size_t)(bb * NH + h) * S + tt) * DH + dh);
      dst[idx] = (_Float16)v;
    }
  }
}

// ---------------- flash attention: one wave per (b, h, qtile16) ----------------
__global__ __launch_bounds__(256) void k_attn(const _Float16* __restrict__ Qh,
                                              const _Float16* __restrict__ Kh,
                                              const _Float16* __restrict__ Vt,
                                              _Float16* __restrict__ ao,
                                              int Sq, int Skv) {
  __shared__ __align__(32) _Float16 pBuf[8 * 16 * 32];
  int tid = threadIdx.x, lane = tid & 31, wave = tid >> 5;
  int gw = blockIdx.x * 8 + wave;
  int qtPerHead = Sq >> 4;
  int b = gw / (NH * qtPerHead);
  int r = gw - b * (NH * qtPerHead);
  int h = r / qtPerHead;
  int qt = r - h * qtPerHead;
  _Float16* myP = pBuf + wave * 512;

  const _Float16* Qb = Qh + ((size_t)(b * NH + h) * Sq + qt * 16) * DH;
  const _Float16* Kbase = Kh + (size_t)(b * NH + h) * Skv * DH;
  const _Float16* Vbase = Vt + (size_t)(b * NH + h) * DH * Skv;
  v16h aq = load_a_frag(Qb, DH, lane);

  v8f o0 = {}, o1 = {};
  float mrow[8], lrow[8];
#pragma unroll
  for (int e = 0; e < 8; ++e) { mrow[e] = -__builtin_inff(); lrow[e] = 0.0f; }
  const float scale = 0.17677669529663687f; // 1/sqrt(32)
  int n = lane & 15, half = lane >> 4;

  for (int c = 0; c < Skv; c += 32) {
    v16h bk0 = load_b_frag(Kbase + (size_t)c * DH, DH, lane);
    v16h bk1 = load_b_frag(Kbase + (size_t)(c + 16) * DH, DH, lane);
    v8f z = {};
    v8f s0 = wmma16(aq, bk0, z);
    v8f s1 = wmma16(aq, bk1, z);
    float p0[8], p1[8], mx[8];
#pragma unroll
    for (int e = 0; e < 8; ++e) {
      p0[e] = s0[e] * scale; p1[e] = s1[e] * scale;
      mx[e] = fmaxf(p0[e], p1[e]);
    }
#pragma unroll
    for (int off = 1; off < 16; off <<= 1)
#pragma unroll
      for (int e = 0; e < 8; ++e) mx[e] = fmaxf(mx[e], __shfl_xor(mx[e], off, 32));
    float rs[8];
#pragma unroll
    for (int e = 0; e < 8; ++e) {
      float mn = fmaxf(mrow[e], mx[e]);
      float al = expf(mrow[e] - mn);
      p0[e] = expf(p0[e] - mn);
      p1[e] = expf(p1[e] - mn);
      mrow[e] = mn;
      rs[e] = p0[e] + p1[e];
      lrow[e] *= al;
      o0[e] *= al; o1[e] *= al;
    }
#pragma unroll
    for (int off = 1; off < 16; off <<= 1)
#pragma unroll
      for (int e = 0; e < 8; ++e) rs[e] += __shfl_xor(rs[e], off, 32);
#pragma unroll
    for (int e = 0; e < 8; ++e) lrow[e] += rs[e];
    // stage P (C-layout -> row-major 16x32) for A-fragment reload
#pragma unroll
    for (int e = 0; e < 8; ++e) {
      int row = e + (half << 3);
      myP[row * 32 + n]      = (_Float16)p0[e];
      myP[row * 32 + 16 + n] = (_Float16)p1[e];
    }
    __syncthreads();
    v16h ap  = load_a_frag(myP, 32, lane);
    v16h bv0 = load_b_frag(Vbase + c, Skv, lane);
    v16h bv1 = load_b_frag(Vbase + (size_t)16 * Skv + c, Skv, lane);
    o0 = wmma16(ap, bv0, o0);
    o1 = wmma16(ap, bv1, o1);
    __syncthreads();
  }
#pragma unroll
  for (int e = 0; e < 8; ++e) {
    float inv = 1.0f / lrow[e];
    int token = qt * 16 + e + (half << 3);
    size_t rowOff = ((size_t)b * Sq + token) * DM + h * DH;
    ao[rowOff + n]      = (_Float16)(o0[e] * inv);
    ao[rowOff + 16 + n] = (_Float16)(o1[e] * inv);
  }
}

// ---------------- generic GEMM + bias (+gelu), f32 or f16 out ----------------
__global__ __launch_bounds__(256) void k_gemm(const _Float16* __restrict__ A,
                                              const _Float16* __restrict__ W,
                                              const float* __restrict__ bias,
                                              float* __restrict__ outF,
                                              _Float16* __restrict__ outH,
                                              int K, int Ntot, int actGelu) {
  int tid = threadIdx.x, lane = tid & 31, wave = tid >> 5;
  int rowBase = blockIdx.x * 32;
  int colBase = blockIdx.y * 128;
  for (int i = 0; i < 2; ++i) {
    int t = wave * 2 + i, mT = t & 1, nT = t >> 1;
    v8f acc = {};
    for (int kc = 0; kc < K; kc += 32) {
      v16h a = load_a_frag(A + (size_t)(rowBase + mT * 16) * K + kc, K, lane);
      v16h b = load_b_frag(W + (size_t)(colBase + nT * 16) * K + kc, K, lane);
      acc = wmma16(a, b, acc);
    }
    int n = lane & 15, half = lane >> 4;
    int col = colBase + nT * 16 + n;
    float bv = bias[col];
#pragma unroll
    for (int e = 0; e < 8; ++e) {
      int m = rowBase + mT * 16 + e + (half << 3);
      float v = acc[e] + bv;
      if (actGelu) v = gelu_f(v);
      if (outH) outH[(size_t)m * Ntot + col] = (_Float16)v;
      else      outF[(size_t)m * Ntot + col] = v;
    }
  }
}

// ---------------- residual + LayerNorm (per token) ----------------
__global__ __launch_bounds__(128) void k_resln(const float* __restrict__ res,
                                               const float* __restrict__ add,
                                               const float* __restrict__ g,
                                               const float* __restrict__ be,
                                               float* __restrict__ x32,
                                               _Float16* __restrict__ x16) {
  __shared__ float red[4];
  int m = blockIdx.x, t = threadIdx.x;
  float v = res[(size_t)m * DM + t] + add[(size_t)m * DM + t];
  float s = v;
#pragma unroll
  for (int o = 16; o > 0; o >>= 1) s += __shfl_xor(s, o, 32);
  if ((t & 31) == 0) red[t >> 5] = s;
  __syncthreads();
  float mean = (red[0] + red[1] + red[2] + red[3]) * (1.0f / DM);
  __syncthreads();
  float d = v - mean, sq = d * d;
#pragma unroll
  for (int o = 16; o > 0; o >>= 1) sq += __shfl_xor(sq, o, 32);
  if ((t & 31) == 0) red[t >> 5] = sq;
  __syncthreads();
  float var = (red[0] + red[1] + red[2] + red[3]) * (1.0f / DM);
  float y = d * rsqrtf(var + LN_EPS) * g[t] + be[t];
  x32[(size_t)m * DM + t] = y;
  x16[(size_t)m * DM + t] = (_Float16)y;
}

// ---------------- host orchestration ----------------
extern "C" void kernel_launch(void* const* d_in, const int* in_sizes, int n_in,
                              void* d_out, int out_size, void* d_ws, size_t ws_size,
                              hipStream_t stream) {
  const float* geom   = (const float*)d_in[0];
  const float* feat_w = (const float*)d_in[2];
  const float* feat_b = (const float*)d_in[3];
  const float* mlp_w1 = (const float*)d_in[4];
  const float* mlp_b1 = (const float*)d_in[5];
  const float* mlp_g1 = (const float*)d_in[6];
  const float* mlp_be1= (const float*)d_in[7];
  const float* mlp_w2 = (const float*)d_in[8];
  const float* mlp_b2 = (const float*)d_in[9];
  const float* mlp_g2 = (const float*)d_in[10];
  const float* mlp_be2= (const float*)d_in[11];
  const float* mlp_w3 = (const float*)d_in[12];
  const float* mlp_b3 = (const float*)d_in[13];
  const float* blk_in_w = (const float*)d_in[14];
  const float* blk_in_b = (const float*)d_in[15];
  const float* blk_ow   = (const float*)d_in[16];
  const float* blk_ob   = (const float*)d_in[17];
  const float* blk_g1   = (const float*)d_in[18];
  const float* blk_be1  = (const float*)d_in[19];
  const float* blk_g2   = (const float*)d_in[20];
  const float* blk_be2  = (const float*)d_in[21];
  const float* blk_f1w  = (const float*)d_in[22];
  const float* blk_f1b  = (const float*)d_in[23];
  const float* blk_f2w  = (const float*)d_in[24];
  const float* blk_f2b  = (const float*)d_in[25];
  const float* k_wm = (const float*)d_in[26];
  const float* k_bv = (const float*)d_in[27];
  const float* v_wm = (const float*)d_in[28];
  const float* v_bv = (const float*)d_in[29];

  char* w = (char*)d_ws;
  size_t off = 0;
  auto take = [&](size_t bytes) -> void* {
    void* p = w + off;
    off = (off + bytes + 255) & ~(size_t)255;
    return p;
  };
  _Float16* gfeat16 = (_Float16*)take((size_t)BB * NN * DM * 2);
  int*      fidx    = (int*)take((size_t)BB * NS * 4);
  float*    cent    = (float*)take((size_t)BB * NS * 2 * 4);
  int*      knn     = (int*)take((size_t)BB * NS * NP * 4);
  int*      cnts    = (int*)take((size_t)BB * NS * 4);
  float*    x32     = (float*)take((size_t)BB * NS * DM * 4);
  _Float16* x16     = (_Float16*)take((size_t)BB * NS * DM * 2);
  _Float16* qh      = (_Float16*)take((size_t)BB * NS * DM * 2);
  _Float16* kh      = (_Float16*)take((size_t)BB * NN * DM * 2);
  _Float16* vt      = (_Float16*)take((size_t)BB * NN * DM * 2);
  _Float16* ao      = (_Float16*)take((size_t)BB * NS * DM * 2);
  float*    tmp32   = (float*)take((size_t)BB * NS * DM * 4);
  _Float16* ff1     = (_Float16*)take((size_t)BB * NS * FF * 2);
  _Float16* w1p  = (_Float16*)take((size_t)MLPH * 160 * 2);
  _Float16* w2h  = (_Float16*)take((size_t)MLPH * MLPH * 2);
  _Float16* w3h  = (_Float16*)take((size_t)DM * MLPH * 2);
  _Float16* inwh = (_Float16*)take((size_t)4 * 3 * DM * DM * 2);
  _Float16* owh  = (_Float16*)take((size_t)4 * DM * DM * 2);
  _Float16* f1wh = (_Float16*)take((size_t)4 * FF * DM * 2);
  _Float16* f2wh = (_Float16*)take((size_t)4 * DM * FF * 2);
  _Float16* kwh  = (_Float16*)take((size_t)DM * DM * 2);
  _Float16* vwh  = (_Float16*)take((size_t)DM * DM * 2);

  auto cvt = [&](const float* s, _Float16* d, int n) {
    k_cvt<<<(n + 255) / 256, 256, 0, stream>>>(s, d, n);
  };
  k_cvt_pad<<<(MLPH * 160 + 255) / 256, 256, 0, stream>>>(mlp_w1, w1p, MLPH, DM + 2, 160);
  cvt(mlp_w2, w2h, MLPH * MLPH);
  cvt(mlp_w3, w3h, DM * MLPH);
  cvt(blk_in_w, inwh, 4 * 3 * DM * DM);
  cvt(blk_ow, owh, 4 * DM * DM);
  cvt(blk_f1w, f1wh, 4 * FF * DM);
  cvt(blk_f2w, f2wh, 4 * DM * FF);
  cvt(k_wm, kwh, DM * DM);
  cvt(v_wm, vwh, DM * DM);

  k_gfeat<<<BB * NN, 128, 0, stream>>>(geom, feat_w, feat_b, gfeat16);
  k_fps<<<BB, 256, 0, stream>>>(geom, fidx, cent);
  k_ballq<<<dim3(NS / 256, BB), 256, 0, stream>>>(geom, cent, knn, cnts);
  k_groupmlp<<<BB * NS, 256, 0, stream>>>(geom, gfeat16, knn, cnts, cent,
                                          w1p, mlp_b1, mlp_g1, mlp_be1,
                                          w2h, mlp_b2, mlp_g2, mlp_be2,
                                          w3h, mlp_b3, x32, x16);

  int Mq = BB * NS; // 8192 query tokens
  for (int i = 0; i < 4; ++i) {
    const _Float16* kvsrc = (i < 2) ? gfeat16 : x16;
    int Skv = (i < 2) ? NN : NS;
    int Mkv = BB * Skv;
    const _Float16* wq = inwh + (size_t)i * 3 * DM * DM;
    const _Float16* wk = wq + DM * DM;
    const _Float16* wv = wk + DM * DM;
    const float* bq = blk_in_b + i * 3 * DM;
    k_proj<<<Mq / 32, 256, 0, stream>>>(x16, wq, bq, qh, NS, 0);
    k_proj<<<Mkv / 32, 256, 0, stream>>>(kvsrc, wk, bq + DM, kh, Skv, 0);
    k_proj<<<Mkv / 32, 256, 0, stream>>>(kvsrc, wv, bq + 2 * DM, vt, Skv, 1);
    k_attn<<<(BB * NH * (NS / 16)) / 8, 256, 0, stream>>>(qh, kh, vt, ao, NS, Skv);
    k_gemm<<<dim3(Mq / 32, 1), 256, 0, stream>>>(ao, owh + (size_t)i * DM * DM,
                                                 blk_ob + i * DM, tmp32, nullptr,
                                                 DM, DM, 0);
    k_resln<<<Mq, 128, 0, stream>>>(x32, tmp32, blk_g1 + i * DM, blk_be1 + i * DM,
                                    x32, x16);
    k_gemm<<<dim3(Mq / 32, FF / 128), 256, 0, stream>>>(x16, f1wh + (size_t)i * FF * DM,
                                                        blk_f1b + i * FF, nullptr, ff1,
                                                        DM, FF, 1);
    k_gemm<<<dim3(Mq / 32, 1), 256, 0, stream>>>(ff1, f2wh + (size_t)i * DM * FF,
                                                 blk_f2b + i * DM, tmp32, nullptr,
                                                 FF, DM, 0);
    k_resln<<<Mq, 128, 0, stream>>>(x32, tmp32, blk_g2 + i * DM, blk_be2 + i * DM,
                                    x32, x16);
  }
  float* outK = (float*)d_out;
  float* outV = outK + (size_t)BB * NS * DM;
  k_gemm<<<dim3(Mq / 32, 1), 256, 0, stream>>>(x16, kwh, k_bv, outK, nullptr, DM, DM, 0);
  k_gemm<<<dim3(Mq / 32, 1), 256, 0, stream>>>(x16, vwh, v_bv, outV, nullptr, DM, DM, 0);
}
